// MultiHeadedAttention_67327907332401
// MI455X (gfx1250) — matrix operand order repaired and checked by reading, not compiled, not run on hardware
//
#include <hip/hip_runtime.h>

typedef __attribute__((ext_vector_type(16))) _Float16 v16h;
typedef __attribute__((ext_vector_type(8)))  float    v8f;
typedef __attribute__((ext_vector_type(4)))  int      v4i;

#define B_  4
#define S_  2048
#define D_  1024
#define H_  16
#define DK_ 64

#if defined(__HIP_DEVICE_COMPILE__) && __has_builtin(__builtin_amdgcn_global_load_async_to_lds_b128)
#define HAVE_ASYNC_LDS 1
#else
#define HAVE_ASYNC_LDS 0
#endif

__device__ __forceinline__ void async_lds_wait0() {
#if defined(__HIP_DEVICE_COMPILE__)
#if __has_builtin(__builtin_amdgcn_s_wait_asynccnt)
    __builtin_amdgcn_s_wait_asynccnt(0);
#else
    asm volatile("s_wait_asynccnt 0x0" ::: "memory");
#endif
#endif
}

#if HAVE_ASYNC_LDS
typedef __attribute__((address_space(1))) v4i* glb_v4i_p;
typedef __attribute__((address_space(3))) v4i* lds_v4i_p;

__device__ __forceinline__ glb_v4i_p as_glb(const void* p) {
    void* p0 = (void*)p;          // strip const
    v4i*  p1 = (v4i*)p0;          // reinterpret pointee
    return (glb_v4i_p)p1;         // addrspacecast generic -> global
}
__device__ __forceinline__ lds_v4i_p as_lds(void* p) {
    v4i* p1 = (v4i*)p;            // reinterpret pointee
    return (lds_v4i_p)p1;         // addrspacecast generic -> LDS
}
#endif

// ------------------------------------------------------------------
// Pack int32 mask (B,S,S) into bitwords: word[(b*S+q)*(S/32)+c] bit i = mask[b][q][c*32+i]
// ------------------------------------------------------------------
__global__ void pack_mask_kernel(const int* __restrict__ mask,
                                 unsigned* __restrict__ bits, int nwords) {
    int idx = blockIdx.x * blockDim.x + threadIdx.x;
    if (idx >= nwords) return;
    const int* p = mask + (size_t)idx * 32;
    unsigned w = 0;
#pragma unroll
    for (int i = 0; i < 32; ++i) w |= (p[i] != 0 ? 1u : 0u) << i;
    bits[idx] = w;
}

// ------------------------------------------------------------------
// Tiled WMMA GEMM: C[M,N] = (A[M,K] @ W[N,K]^T + bias[N]) * outScale
// MODE 0: f16 out, scatter to [B,H,S,DK]     (Q, K projections)
// MODE 1: f16 out, scatter to [B,H,DK,S]     (V projection, transposed)
// MODE 2: fp32 out, row-major [M,N]          (final output projection)
// Block tile 128x128, K-tile 32, 256 threads (8 waves), wave tile 32x64.
// ------------------------------------------------------------------
template <typename AT, int MODE>
__global__ __launch_bounds__(256)
void gemm_wmma_kernel(const AT* __restrict__ A, const float* __restrict__ W,
                      const float* __restrict__ bias, void* __restrict__ outp,
                      int M, int N, int K, float outScale) {
    const int BK = 32;
    __shared__ alignas(16) _Float16 Alds[128][BK + 8];
    __shared__ alignas(16) _Float16 Wlds[128][BK + 8];

    const int t    = threadIdx.x;
    const int lane = t & 31;
    const int wave = t >> 5;
    const int half = lane >> 4;
    const int ln   = lane & 15;
    const int waveM = (wave >> 1) * 32;
    const int waveN = (wave & 1) * 64;
    const int blockN = blockIdx.x * 128;
    const int blockM = blockIdx.y * 128;

    v8f acc[2][4];
#pragma unroll
    for (int i = 0; i < 2; ++i)
#pragma unroll
        for (int j = 0; j < 4; ++j)
#pragma unroll
            for (int r = 0; r < 8; ++r) acc[i][j][r] = 0.0f;

    const int ldRow = t >> 1;
    const int ldCol = (t & 1) * 16;

    for (int k0 = 0; k0 < K; k0 += BK) {
        // Stage A-tile (128x32) and W-tile (128x32) -> f16 in LDS
        {
            const AT* ap = A + (size_t)(blockM + ldRow) * K + k0 + ldCol;
#pragma unroll
            for (int i = 0; i < 16; ++i) Alds[ldRow][ldCol + i] = (_Float16)ap[i];
            const float* wp = W + (size_t)(blockN + ldRow) * K + k0 + ldCol;
#pragma unroll
            for (int i = 0; i < 16; ++i) Wlds[ldRow][ldCol + i] = (_Float16)wp[i];
            if (k0 + BK < K) {                    // prefetch next k-tile
                __builtin_prefetch(ap + BK, 0, 0);
                __builtin_prefetch(wp + BK, 0, 0);
            }
        }
        __syncthreads();

        v16h af[2], bf[4];
#pragma unroll
        for (int i = 0; i < 2; ++i) {
#pragma unroll
            for (int v = 0; v < 8; ++v) {
                int kk = (v < 4 ? 0 : 16) + half * 8 + 2 * (v & 3); // 16-bit A layout
                af[i][2 * v]     = Alds[waveM + i * 16 + ln][kk];
                af[i][2 * v + 1] = Alds[waveM + i * 16 + ln][kk + 1];
            }
        }
#pragma unroll
        for (int j = 0; j < 4; ++j) {
#pragma unroll
            for (int v = 0; v < 8; ++v) {
                int kk = (v < 4 ? 0 : 16) + half * 8 + 2 * (v & 3); // 16-bit B layout
                bf[j][2 * v]     = Wlds[waveN + j * 16 + ln][kk];
                bf[j][2 * v + 1] = Wlds[waveN + j * 16 + ln][kk + 1];
            }
        }
#pragma unroll
        for (int i = 0; i < 2; ++i)
#pragma unroll
            for (int j = 0; j < 4; ++j)
                acc[i][j] = __builtin_amdgcn_wmma_f32_16x16x32_f16(
                    false, af[i], false, bf[j], (short)0, acc[i][j], false, false);
        __syncthreads();
    }

    // Epilogue: bias add + optional scale + mode-dependent scatter
#pragma unroll
    for (int i = 0; i < 2; ++i) {
#pragma unroll
        for (int j = 0; j < 4; ++j) {
#pragma unroll
            for (int r = 0; r < 8; ++r) {
                int row = blockM + waveM + i * 16 + r + 8 * half;
                int col = blockN + waveN + j * 16 + ln;
                float val = (acc[i][j][r] + bias[col]) * outScale;
                if (MODE == 2) {
                    ((float*)outp)[(size_t)row * N + col] = val;
                } else {
                    int s  = row & (S_ - 1);
                    int bb = row / S_;
                    int hh = col / DK_;
                    int dd = col & (DK_ - 1);
                    _Float16* o = (_Float16*)outp;
                    if (MODE == 0)
                        o[(((size_t)bb * H_ + hh) * S_ + s) * DK_ + dd] = (_Float16)val;
                    else
                        o[(((size_t)bb * H_ + hh) * DK_ + dd) * S_ + s] = (_Float16)val;
                }
            }
        }
    }
}

// ------------------------------------------------------------------
// Flash attention: 4 waves/block, each wave owns a 16-query tile; the block
// shares LDS-staged 32-key K/V chunks (async global->LDS on gfx1250).
// Q is pre-scaled by 1/sqrt(DK) at projection time.
// ------------------------------------------------------------------
__global__ __launch_bounds__(128)
void attn_kernel(const _Float16* __restrict__ Qh, const _Float16* __restrict__ Kh,
                 const _Float16* __restrict__ Vt, const unsigned* __restrict__ mbits,
                 _Float16* __restrict__ AO) {
    __shared__ alignas(16) _Float16 Klds[32][DK_ + 8];   // pitch 144B (16B multiple)
    __shared__ alignas(16) _Float16 Vlds[DK_][32 + 8];   // pitch 80B  (16B multiple)
    __shared__ alignas(16) _Float16 Plds[4][16][32 + 8];

    const int t    = threadIdx.x;
    const int lane = t & 31;
    const int wave = t >> 5;
    const int half = lane >> 4;
    const int ln   = lane & 15;

    const int bh = blockIdx.x;
    const int b  = bh / H_;
    const int h  = bh % H_;
    const int q0 = blockIdx.y * 64 + wave * 16;

    const size_t headOff = (size_t)bh * S_ * DK_;

    // per-thread staging coordinates (32 bytes of K and 32 bytes of V each)
    const int skey = t >> 2, sdkb = (t & 3) * 16;
    const int sdk  = t >> 1, skb  = (t & 1) * 16;

    // Q fragments (A layout), dk 0..31 and 32..63
    v16h qf[2];
    {
        const _Float16* qp = Qh + headOff + (size_t)(q0 + ln) * DK_;
#pragma unroll
        for (int st = 0; st < 2; ++st)
#pragma unroll
            for (int v = 0; v < 8; ++v) {
                int kk = st * 32 + (v < 4 ? 0 : 16) + half * 8 + 2 * (v & 3);
                qf[st][2 * v]     = qp[kk];
                qf[st][2 * v + 1] = qp[kk + 1];
            }
    }

    v8f acc[4];
#pragma unroll
    for (int j = 0; j < 4; ++j)
#pragma unroll
        for (int r = 0; r < 8; ++r) acc[j][r] = 0.0f;
    float mrow[8], lrow[8];
#pragma unroll
    for (int r = 0; r < 8; ++r) { mrow[r] = -3.0e38f; lrow[r] = 0.0f; }

    for (int chunk = 0; chunk < S_ / 32; ++chunk) {
        const int k0 = chunk * 32;
        __syncthreads();    // previous chunk's consumers done before overwrite
        {
            const _Float16* kp = Kh + headOff + (size_t)(k0 + skey) * DK_ + sdkb;
            const _Float16* vp = Vt + headOff + (size_t)sdk * S_ + k0 + skb;
#if HAVE_ASYNC_LDS
            __builtin_amdgcn_global_load_async_to_lds_b128(
                as_glb(kp), as_lds(&Klds[skey][sdkb]), 0, 0);
            __builtin_amdgcn_global_load_async_to_lds_b128(
                as_glb(kp), as_lds(&Klds[skey][sdkb]), 16, 0);
            __builtin_amdgcn_global_load_async_to_lds_b128(
                as_glb(vp), as_lds(&Vlds[sdk][skb]), 0, 0);
            __builtin_amdgcn_global_load_async_to_lds_b128(
                as_glb(vp), as_lds(&Vlds[sdk][skb]), 16, 0);
#else
#pragma unroll
            for (int i = 0; i < 16; ++i) Klds[skey][sdkb + i] = kp[i];
#pragma unroll
            for (int i = 0; i < 16; ++i) Vlds[sdk][skb + i] = vp[i];
#endif
            if (k0 + 32 < S_) {                 // prefetch next chunk
                __builtin_prefetch(kp + 32 * DK_, 0, 0);
                __builtin_prefetch(vp + 32, 0, 0);
            }
        }
#if HAVE_ASYNC_LDS
        async_lds_wait0();
#endif
        __syncthreads();

        // Scores: S(16x32) = Q(16x64) x K^T(64x32); 2 n-tiles x 2 k-steps
        v8f sc[2];
#pragma unroll
        for (int j = 0; j < 2; ++j) {
#pragma unroll
            for (int r = 0; r < 8; ++r) sc[j][r] = 0.0f;
#pragma unroll
            for (int st = 0; st < 2; ++st) {
                v16h bf;
#pragma unroll
                for (int v = 0; v < 8; ++v) {
                    int kk = st * 32 + (v < 4 ? 0 : 16) + half * 8 + 2 * (v & 3);
                    bf[2 * v]     = Klds[j * 16 + ln][kk];
                    bf[2 * v + 1] = Klds[j * 16 + ln][kk + 1];
                }
                sc[j] = __builtin_amdgcn_wmma_f32_16x16x32_f16(
                    false, qf[st], false, bf, (short)0, sc[j], false, false);
            }
        }

        // Mask + online softmax (rows r+8*half, cols = key within chunk)
        float p0[8], p1[8];
#pragma unroll
        for (int r = 0; r < 8; ++r) {
            unsigned mw = mbits[((size_t)b * S_ + q0 + r + 8 * half) * (S_ / 32) + chunk];
            float s0 = ((mw >> ln) & 1u)        ? sc[0][r] : -1.0e9f;
            float s1 = ((mw >> (16 + ln)) & 1u) ? sc[1][r] : -1.0e9f;
            float v = fmaxf(s0, s1);
#pragma unroll
            for (int off = 1; off < 16; off <<= 1) v = fmaxf(v, __shfl_xor(v, off));
            float mnew  = fmaxf(mrow[r], v);
            float scale = __expf(mrow[r] - mnew);
            mrow[r] = mnew;
            p0[r] = __expf(s0 - mnew);
            p1[r] = __expf(s1 - mnew);
            float ps = p0[r] + p1[r];
#pragma unroll
            for (int off = 1; off < 16; off <<= 1) ps += __shfl_xor(ps, off);
            lrow[r] = lrow[r] * scale + ps;
#pragma unroll
            for (int j = 0; j < 4; ++j) acc[j][r] *= scale;
        }

        // P (C layout) -> LDS row-major -> reread as A fragment (wave-local)
#pragma unroll
        for (int r = 0; r < 8; ++r) {
            Plds[wave][r + 8 * half][ln]      = (_Float16)p0[r];
            Plds[wave][r + 8 * half][16 + ln] = (_Float16)p1[r];
        }
        v16h pf;
#pragma unroll
        for (int v = 0; v < 8; ++v) {
            int kk = (v < 4 ? 0 : 16) + half * 8 + 2 * (v & 3);
            pf[2 * v]     = Plds[wave][ln][kk];
            pf[2 * v + 1] = Plds[wave][ln][kk + 1];
        }

        // O += P(16x32) x V(32x64); 4 n-tiles
#pragma unroll
        for (int j = 0; j < 4; ++j) {
            v16h bf;
#pragma unroll
            for (int v = 0; v < 8; ++v) {
                int kk = (v < 4 ? 0 : 16) + half * 8 + 2 * (v & 3);
                bf[2 * v]     = Vlds[j * 16 + ln][kk];
                bf[2 * v + 1] = Vlds[j * 16 + ln][kk + 1];
            }
            acc[j] = __builtin_amdgcn_wmma_f32_16x16x32_f16(
                false, pf, false, bf, (short)0, acc[j], false, false);
        }
    }

    // Normalize and write attention output in (B,S,H,DK) == (B*S, D) layout
#pragma unroll
    for (int j = 0; j < 4; ++j)
#pragma unroll
        for (int r = 0; r < 8; ++r) {
            int q  = q0 + r + 8 * half;
            int dk = j * 16 + ln;
            float o = acc[j][r] / lrow[r];
            AO[(((size_t)b * S_ + q) * H_ + h) * DK_ + dk] = (_Float16)o;
        }
}

// ------------------------------------------------------------------
extern "C" void kernel_launch(void* const* d_in, const int* in_sizes, int n_in,
                              void* d_out, int out_size, void* d_ws, size_t ws_size,
                              hipStream_t stream) {
    const float* query = (const float*)d_in[0];
    const float* key   = (const float*)d_in[1];
    const float* value = (const float*)d_in[2];
    const int*   mask  = (const int*)d_in[3];
    const float* Wq = (const float*)d_in[4];
    const float* bq = (const float*)d_in[5];
    const float* Wk = (const float*)d_in[6];
    const float* bk = (const float*)d_in[7];
    const float* Wv = (const float*)d_in[8];
    const float* bv = (const float*)d_in[9];
    const float* Wo = (const float*)d_in[10];
    const float* bo = (const float*)d_in[11];
    float* out = (float*)d_out;

    const size_t elems = (size_t)B_ * S_ * D_;       // 8,388,608
    char* ws = (char*)d_ws;
    _Float16* Qh = (_Float16*)(ws);
    _Float16* Kh = (_Float16*)(ws + 2 * elems);
    _Float16* Vt = (_Float16*)(ws + 4 * elems);
    _Float16* AO = (_Float16*)(ws + 6 * elems);
    unsigned* mb = (unsigned*)(ws + 8 * elems);

    const int nwords = B_ * S_ * (S_ / 32);
    pack_mask_kernel<<<(nwords + 255) / 256, 256, 0, stream>>>(mask, mb, nwords);

    dim3 ggrid(D_ / 128, (B_ * S_) / 128);
    // Q projection pre-scaled by 1/sqrt(DK) = 0.125
    gemm_wmma_kernel<float, 0><<<ggrid, 256, 0, stream>>>(query, Wq, bq, Qh, B_ * S_, D_, D_, 0.125f);
    gemm_wmma_kernel<float, 0><<<ggrid, 256, 0, stream>>>(key,   Wk, bk, Kh, B_ * S_, D_, D_, 1.0f);
    gemm_wmma_kernel<float, 1><<<ggrid, 256, 0, stream>>>(value, Wv, bv, Vt, B_ * S_, D_, D_, 1.0f);

    attn_kernel<<<dim3(B_ * H_, S_ / 64), 128, 0, stream>>>(Qh, Kh, Vt, mb, AO);

    gemm_wmma_kernel<_Float16, 2><<<ggrid, 256, 0, stream>>>(AO, Wo, bo, out, B_ * S_, D_, D_, 1.0f);
}
